// PSA_12799002542332
// MI455X (gfx1250) — matrix operand order repaired and checked by reading, not compiled, not run on hardware
//
#include <hip/hip_runtime.h>
#include <hip/hip_bf16.h>

// ---------------------------------------------------------------------------
// PSA (permuted window attention) fused for MI455X / gfx1250, wave32 + WMMA.
//   prep : weights fp32->bf16, bias table gather -> dense (H,256,64) f32
//   main : per-window fused  x->LDS(bf16) -> KV proj(scatter) -> per-head
//          [Q_h proj -> QK^T(+bias C) -> softmax(in-reg, shfl_xor) -> PV] -> O(bf16)
//   proj : O @ Wproj^T + bproj -> f32 out (async LDS staging)
// All matmuls: v_wmma_f32_16x16x32_bf16, dual-tile chains for load/WMMA overlap.
// ---------------------------------------------------------------------------

typedef __attribute__((ext_vector_type(16))) __bf16 bf16x16;
typedef __attribute__((ext_vector_type(8)))  float  f32x8;

#define NWIN 1024
#define NQ   256
#define CDIM 256
#define NKV  64
#define HDS  8
#define HD   32
#define XS_S 264   // x / O staging stride (bf16 elems), conflict-free b128 rows
#define KS_S 40    // K per-head row stride (64 rows x 32 cols)
#define VS_S 72    // V^T per-head row stride (32 rows x 64 cols)
#define QH_S 40    // Q_h stride (256 rows x 32 cols)
#define AT_S 72    // attn stride (256 rows x 64 cols)
#define QSCALE 0.17677669529663689f  // 32^-0.5

union FragU { bf16x16 v; uint4 u[2]; };

// packed f32x2 -> bf16x2 (RNE) via gfx1250 VALU
__device__ __forceinline__ unsigned cvt_pk_bf16(float a, float b) {
  unsigned r;
  asm("v_cvt_pk_bf16_f32 %0, %1, %2" : "=v"(r) : "v"(a), "v"(b));
  return r;
}
__device__ __forceinline__ unsigned short f2bf_bits(float f) {
  return (unsigned short)(cvt_pk_bf16(f, 0.0f) & 0xffffu);
}
__device__ __forceinline__ __bf16 f2bf(float f) {
  return __builtin_bit_cast(__bf16, f2bf_bits(f));
}

// A fragment (16x32): per-lane 16 contiguous bf16 (one row's K-half, row-major src)
__device__ __forceinline__ bf16x16 ld_fragA(const __bf16* p) {
  FragU f;
  f.u[0] = *(const uint4*)p;
  f.u[1] = *(const uint4*)(p + 8);
  return f.v;
}
// B fragment (32x16): per-lane 16 contiguous bf16 (one column's K-half)
__device__ __forceinline__ bf16x16 ld_fragB(const __bf16* p) {
  FragU f;
  f.u[0] = *(const uint4*)p;
  f.u[1] = *(const uint4*)(p + 8);
  return f.v;
}

__device__ __forceinline__ f32x8 wmma_bf16(bf16x16 a, bf16x16 b, f32x8 c) {
  return __builtin_amdgcn_wmma_f32_16x16x32_bf16(false, a, false, b, (short)0, c, false, false);
}

// ---------------------------------------------------------------------------
// prep: fp32->bf16 weight conversion + dense bias expansion
// ---------------------------------------------------------------------------
__global__ __launch_bounds__(256) void psa_prep(
    const float* __restrict__ Wq, const float* __restrict__ Wkv,
    const float* __restrict__ Wproj,
    const float* __restrict__ bias_table, const int* __restrict__ rel_index,
    unsigned short* __restrict__ Wqb, unsigned short* __restrict__ Wkvb,
    unsigned short* __restrict__ Wprojb, float* __restrict__ biasx)
{
  int i = blockIdx.x * 256 + threadIdx.x;
  if (i < 65536)       Wqb[i]           = f2bf_bits(Wq[i]);
  else if (i < 98304)  Wkvb[i - 65536]  = f2bf_bits(Wkv[i - 65536]);
  else if (i < 163840) Wprojb[i - 98304]= f2bf_bits(Wproj[i - 98304]);
  else if (i < 294912) {
    int j  = i - 163840;
    int h  = j >> 14;        // 16384 = 256*64 per head
    int qk = j & 16383;
    biasx[j] = bias_table[rel_index[qk] * 8 + h];
  }
}

// ---------------------------------------------------------------------------
// main fused kernel: one block per window
// ---------------------------------------------------------------------------
__global__ __launch_bounds__(256) void psa_main(
    const float* __restrict__ x,
    const float* __restrict__ bq,
    const float* __restrict__ bkv,
    const __bf16* __restrict__ Wqb,
    const __bf16* __restrict__ Wkvb,
    const float* __restrict__ biasx,
    unsigned short* __restrict__ Og)
{
  extern __shared__ char smem[];
  __bf16* xs = (__bf16*)smem;                 // 256 x 264
  __bf16* ks = xs + NQ * XS_S;                // 8 heads x (64 x 40)
  __bf16* vs = ks + HDS * NKV * KS_S;         // 8 heads x (32 x 72)  (V^T)
  __bf16* qh = vs + HDS * HD * VS_S;          // 256 x 40
  __bf16* at = qh + NQ * QH_S;                // 256 x 72

  const int b    = blockIdx.x;
  const int tid  = threadIdx.x;
  const int wv   = __builtin_amdgcn_readfirstlane(tid >> 5);  // uniform wave id
  const int lane = tid & 31;
  const int lrow = lane & 15;
  const int lseg = lane >> 4;
  const int koffA = lseg * 16;

  // ---- Phase 1: stage x window -> LDS bf16 (packed cvt) ----
  const float* xw = x + (size_t)b * (NQ * CDIM);
  #pragma unroll 4
  for (int it = 0; it < 64; ++it) {
    int linear = it * 1024 + tid * 4;
    int row = linear >> 8, col = linear & 255;
    float4 f = *(const float4*)(xw + linear);
    unsigned lo = cvt_pk_bf16(f.x, f.y);
    unsigned hi = cvt_pk_bf16(f.z, f.w);
    *(uint2*)(xs + row * XS_S + col) = make_uint2(lo, hi);
  }
  __syncthreads();

  // ---- Phase 2: KV = x @ Wkv^T + bkv, pixel-unshuffle scatter to K / V^T ----
  {
    const int tn = wv;                       // each wave owns one 16-col strip
    const __bf16* brow = Wkvb + (tn * 16 + lrow) * CDIM + lseg * 16;
    bf16x16 bfr[8];
    #pragma unroll
    for (int kk = 0; kk < 8; ++kk) bfr[kk] = ld_fragB(brow + kk * 32);

    const int s  = (tn < 4) ? 0 : 1;         // 0=K half, 1=V half (uniform)
    const int dh = (tn >> 1) & 1;            // d bit5 (uniform)
    const int e  = (tn & 1) * 16 + lrow;     // head-dim element
    const float bv = bkv[tn * 16 + lrow];

    for (int tm2 = 0; tm2 < 8; ++tm2) {      // two row strips per iter (dual chain)
      const __bf16* arow0 = xs + (tm2 * 32 + lrow) * XS_S + koffA;
      const __bf16* arow1 = arow0 + 16 * XS_S;
      f32x8 acc0 = {}, acc1 = {};
      #pragma unroll
      for (int kk = 0; kk < 8; ++kk) {
        bf16x16 a0 = ld_fragA(arow0 + kk * 32);
        bf16x16 a1 = ld_fragA(arow1 + kk * 32);
        acc0 = wmma_bf16(a0, bfr[kk], acc0);
        acc1 = wmma_bf16(a1, bfr[kk], acc1);
      }
      #pragma unroll
      for (int half = 0; half < 2; ++half) {
        const f32x8& acc = half ? acc1 : acc0;
        #pragma unroll
        for (int j = 0; j < 8; ++j) {
          int M  = tm2 * 32 + half * 16 + j + (lseg << 3);   // window position n
          int i0 = M >> 5, a2 = (M >> 4) & 1, i1 = (M >> 1) & 7, b2 = M & 1;
          int kvpos = i0 * 8 + i1;
          int h = a2 * 4 + b2 * 2 + dh;
          __bf16 val = f2bf(acc[j] + bv);
          if (s == 0) ks[h * (NKV * KS_S) + kvpos * KS_S + e] = val;  // K[h][kv][e]
          else        vs[h * (HD * VS_S) + e * VS_S + kvpos] = val;   // V^T[h][e][kv]
        }
      }
    }
  }
  __syncthreads();

  // ---- Phase 3: per-head attention ----
  for (int h = 0; h < HDS; ++h) {
    // 3a: Q_h = (x @ Wq_h^T + bq_h) * scale -> qh (256x32 bf16)
    #pragma unroll
    for (int tmi = 0; tmi < 2; ++tmi) {
      int tm = wv * 2 + tmi;
      const __bf16* arow = xs + (tm * 16 + lrow) * XS_S + koffA;
      bf16x16 af[8];
      #pragma unroll
      for (int kk = 0; kk < 8; ++kk) af[kk] = ld_fragA(arow + kk * 32);

      const __bf16* brow0 = Wqb + (h * 32 + lrow) * CDIM + lseg * 16;       // tn=0
      const __bf16* brow1 = brow0 + 16 * CDIM;                              // tn=1
      f32x8 acc0 = {}, acc1 = {};
      #pragma unroll
      for (int kk = 0; kk < 8; ++kk) {
        bf16x16 b0 = ld_fragB(brow0 + kk * 32);
        bf16x16 b1 = ld_fragB(brow1 + kk * 32);
        acc0 = wmma_bf16(af[kk], b0, acc0);
        acc1 = wmma_bf16(af[kk], b1, acc1);
      }
      float bv0 = bq[h * 32 + lrow];
      float bv1 = bq[h * 32 + 16 + lrow];
      #pragma unroll
      for (int j = 0; j < 8; ++j) {
        int M = tm * 16 + j + (lseg << 3);
        qh[M * QH_S + lrow]      = f2bf((acc0[j] + bv0) * QSCALE);
        qh[M * QH_S + 16 + lrow] = f2bf((acc1[j] + bv1) * QSCALE);
      }
    }
    __syncthreads();

    // 3b: attn = Qh @ Kh^T (bias preloaded as WMMA C), softmax in-register
    #pragma unroll
    for (int st = 0; st < 2; ++st) {
      int sm = wv + st * 8;                              // query strip 0..15
      bf16x16 afrag = ld_fragA(qh + (sm * 16 + lrow) * QH_S + koffA);
      bf16x16 bk[4];
      f32x8 accs[4];
      #pragma unroll
      for (int kt = 0; kt < 4; ++kt)
        bk[kt] = ld_fragB(ks + h * (NKV * KS_S) + (kt * 16 + lrow) * KS_S + lseg * 16);
      #pragma unroll
      for (int kt = 0; kt < 4; ++kt) {
        int kv = kt * 16 + lrow;
        f32x8 bi;
        #pragma unroll
        for (int j = 0; j < 8; ++j) {
          int M = sm * 16 + j + (lseg << 3);
          bi[j] = biasx[h * 16384 + M * 64 + kv];
        }
        accs[kt] = wmma_bf16(afrag, bk[kt], bi);
      }
      // softmax over 64 kv: row M lives in one 16-lane half across 4 accs
      #pragma unroll
      for (int j = 0; j < 8; ++j) {
        float mx = fmaxf(fmaxf(accs[0][j], accs[1][j]), fmaxf(accs[2][j], accs[3][j]));
        #pragma unroll
        for (int m = 1; m < 16; m <<= 1) mx = fmaxf(mx, __shfl_xor(mx, m, 32));
        float p0 = __expf(accs[0][j] - mx);
        float p1 = __expf(accs[1][j] - mx);
        float p2 = __expf(accs[2][j] - mx);
        float p3 = __expf(accs[3][j] - mx);
        float sum = (p0 + p1) + (p2 + p3);
        #pragma unroll
        for (int m = 1; m < 16; m <<= 1) sum += __shfl_xor(sum, m, 32);
        float inv = 1.0f / sum;
        int M = sm * 16 + j + (lseg << 3);
        __bf16* arow2 = at + M * AT_S + lrow;
        arow2[0]  = f2bf(p0 * inv);
        arow2[16] = f2bf(p1 * inv);
        arow2[32] = f2bf(p2 * inv);
        arow2[48] = f2bf(p3 * inv);
      }
    }
    __syncthreads();

    // 3c: out_h = attn @ V_h -> global O (bf16), dual-tn chains per strip
    #pragma unroll
    for (int tmi = 0; tmi < 2; ++tmi) {
      int tm = wv * 2 + tmi;
      const __bf16* arow  = at + (tm * 16 + lrow) * AT_S + koffA;
      const __bf16* brow0 = vs + h * (HD * VS_S) + lrow * VS_S + lseg * 16;       // tn=0
      const __bf16* brow1 = brow0 + 16 * VS_S;                                     // tn=1
      bf16x16 a0 = ld_fragA(arow);
      bf16x16 a1 = ld_fragA(arow + 32);
      bf16x16 b00 = ld_fragB(brow0);
      bf16x16 b01 = ld_fragB(brow0 + 32);
      bf16x16 b10 = ld_fragB(brow1);
      bf16x16 b11 = ld_fragB(brow1 + 32);
      f32x8 acc0 = {}, acc1 = {};
      acc0 = wmma_bf16(a0, b00, acc0);
      acc1 = wmma_bf16(a0, b10, acc1);
      acc0 = wmma_bf16(a1, b01, acc0);
      acc1 = wmma_bf16(a1, b11, acc1);
      #pragma unroll
      for (int j = 0; j < 8; ++j) {
        int M = tm * 16 + j + (lseg << 3);
        size_t base = ((size_t)b * NQ + M) * CDIM + h * 32;
        Og[base + lrow]      = f2bf_bits(acc0[j]);
        Og[base + 16 + lrow] = f2bf_bits(acc1[j]);
      }
    }
    __syncthreads();
  }
}

// ---------------------------------------------------------------------------
// proj: out = O @ Wproj^T + bproj  (f32 output); O staged via async LDS loads
// ---------------------------------------------------------------------------
__global__ __launch_bounds__(256) void psa_proj(
    const unsigned short* __restrict__ Og,
    const __bf16* __restrict__ Wprojb,
    const float* __restrict__ bproj,
    float* __restrict__ out)
{
  extern __shared__ char smem[];
  __bf16* os = (__bf16*)smem;                  // 256 x 264
  const int b    = blockIdx.x;
  const int tid  = threadIdx.x;
  const int wv   = __builtin_amdgcn_readfirstlane(tid >> 5);
  const int lane = tid & 31;
  const int lrow = lane & 15;
  const int lseg = lane >> 4;
  const int koffA = lseg * 16;

  // async copy O window (bf16) -> LDS, 16B per lane per issue (ASYNCcnt path)
  const unsigned short* ow = Og + (size_t)b * (NQ * CDIM);
  #pragma unroll 4
  for (int it = 0; it < 32; ++it) {
    int linear = it * 2048 + tid * 8;               // elements (x8 bf16)
    int row = linear >> 8, col = linear & 255;
    unsigned lds_off = (unsigned)(row * XS_S + col) * 2u;
    unsigned goff    = (unsigned)linear * 2u;
    asm volatile("global_load_async_to_lds_b128 %0, %1, %2"
                 :: "v"(lds_off), "v"(goff), "s"(ow) : "memory");
  }
  asm volatile("s_wait_asynccnt 0" ::: "memory");
  __syncthreads();

  #pragma unroll
  for (int tmi = 0; tmi < 2; ++tmi) {
    int tm = wv * 2 + tmi;
    const __bf16* arow = os + (tm * 16 + lrow) * XS_S + koffA;
    bf16x16 af[8];
    #pragma unroll
    for (int kk = 0; kk < 8; ++kk) af[kk] = ld_fragA(arow + kk * 32);

    for (int tn2 = 0; tn2 < 8; ++tn2) {            // two col tiles per iter
      const __bf16* brow0 = Wprojb + (tn2 * 32 + lrow) * CDIM + lseg * 16;
      const __bf16* brow1 = brow0 + 16 * CDIM;
      f32x8 acc0 = {}, acc1 = {};
      #pragma unroll
      for (int kk = 0; kk < 8; ++kk) {
        bf16x16 b0 = ld_fragB(brow0 + kk * 32);
        bf16x16 b1 = ld_fragB(brow1 + kk * 32);
        acc0 = wmma_bf16(af[kk], b0, acc0);
        acc1 = wmma_bf16(af[kk], b1, acc1);
      }
      int col0 = tn2 * 32 + lrow;
      float bp0 = bproj[col0];
      float bp1 = bproj[col0 + 16];
      #pragma unroll
      for (int j = 0; j < 8; ++j) {
        int M = tm * 16 + j + (lseg << 3);
        size_t base = ((size_t)b * NQ + M) * CDIM + col0;
        out[base]      = acc0[j] + bp0;
        out[base + 16] = acc1[j] + bp1;
      }
    }
  }
}

// ---------------------------------------------------------------------------
extern "C" void kernel_launch(void* const* d_in, const int* in_sizes, int n_in,
                              void* d_out, int out_size, void* d_ws, size_t ws_size,
                              hipStream_t stream)
{
  (void)in_sizes; (void)n_in; (void)out_size; (void)ws_size;
  const float* x          = (const float*)d_in[0];
  const float* Wq         = (const float*)d_in[1];
  const float* bq         = (const float*)d_in[2];
  const float* Wkv        = (const float*)d_in[3];
  const float* bkv        = (const float*)d_in[4];
  const float* Wproj      = (const float*)d_in[5];
  const float* bproj      = (const float*)d_in[6];
  const float* bias_table = (const float*)d_in[7];
  const int*   rel_index  = (const int*)d_in[8];

  // workspace layout (bytes):
  //   0        Wqb    bf16 256x256   (131072)
  //   131072   Wkvb   bf16 128x256   ( 65536)
  //   196608   Wprojb bf16 256x256   (131072)
  //   327680   biasx  f32  8x256x64  (524288)
  //   851968   Og     bf16 1024x256x256 (134217728)
  char* ws = (char*)d_ws;
  unsigned short* Wqb    = (unsigned short*)(ws + 0);
  unsigned short* Wkvb   = (unsigned short*)(ws + 131072);
  unsigned short* Wprojb = (unsigned short*)(ws + 196608);
  float*          biasx  = (float*)(ws + 327680);
  unsigned short* Og     = (unsigned short*)(ws + 851968);

  psa_prep<<<1152, 256, 0, stream>>>(Wq, Wkv, Wproj, bias_table, rel_index,
                                     Wqb, Wkvb, Wprojb, biasx);

  size_t lds_main = (size_t)(NQ * XS_S + HDS * NKV * KS_S + HDS * HD * VS_S +
                             NQ * QH_S + NQ * AT_S) * 2;   // 270336 B of 320 KB
  psa_main<<<NWIN, 256, lds_main, stream>>>(
      x, bq, bkv, (const __bf16*)Wqb, (const __bf16*)Wkvb, biasx, Og);

  size_t lds_proj = (size_t)(NQ * XS_S) * 2;               // 135168 B
  psa_proj<<<NWIN, 256, lds_proj, stream>>>(
      Og, (const __bf16*)Wprojb, bproj, (float*)d_out);
}
